// PointNetPlusPlus_16870631538823
// MI455X (gfx1250) — compile-verified, matrix operand order checked by reading, not run on hardware
//
#include <hip/hip_runtime.h>

typedef _Float16 half_t;
typedef __attribute__((ext_vector_type(16))) _Float16 v16h;
typedef __attribute__((ext_vector_type(8)))  _Float16 v8h;
typedef __attribute__((ext_vector_type(8)))  float    v8f;

// ---------------------------------------------------------------------------
// conversion / weight packing
// ---------------------------------------------------------------------------
__global__ void k_f32_to_f16(const float* __restrict__ in, half_t* __restrict__ out, int n) {
  int i = blockIdx.x * blockDim.x + threadIdx.x;
  if (i < n) out[i] = (half_t)in[i];
}

// W is K x N (row-major, f32). Pack as Bt = N x Kpad (f16, transposed, zero padded)
// so WMMA B fragments load as two contiguous 16B chunks per lane.
// scale/shift fold BatchNorm (eval: mean 0 var 1):  out = acc*scale[n] + shift[n]
__global__ void k_pack_linbn(const float* __restrict__ W, const float* __restrict__ b,
                             const float* __restrict__ g, const float* __restrict__ be,
                             int K, int N, int Kpad, int NoutPad, int isBN,
                             half_t* __restrict__ Bt, float* __restrict__ scale,
                             float* __restrict__ shift) {
  int i = blockIdx.x * blockDim.x + threadIdx.x;
  int total = NoutPad * Kpad;
  if (i < total) {
    int n = i / Kpad, k = i - n * Kpad;
    float w = (n < N && k < K) ? W[k * N + n] : 0.0f;
    Bt[i] = (half_t)w;
  }
  if (i < NoutPad) {
    if (i < N) {
      float s = isBN ? g[i] * rsqrtf(1.0f + 1e-5f) : 1.0f;
      scale[i] = s;
      shift[i] = isBN ? (b[i] * s + be[i]) : b[i];
    } else { scale[i] = 0.0f; shift[i] = 0.0f; }
  }
}

// ---------------------------------------------------------------------------
// farthest point sampling (matches reference: start at 0, argmax with first-tie)
// single workgroup, 1024 threads; dist array lives in global scratch
// ---------------------------------------------------------------------------
__global__ __launch_bounds__(1024)
void k_fps(const float* __restrict__ pos, int Np, int nOut,
           int* __restrict__ idxOut, float* __restrict__ dist) {
  __shared__ float svals[32];
  __shared__ int   sidx[32];
  __shared__ int   sFar;
  int t = threadIdx.x;
  for (int j = t; j < Np; j += 1024) dist[j] = 3.4e38f;
  int far = 0;
  __syncthreads();
  for (int i = 0; i < nOut; ++i) {
    if (t == 0) idxOut[i] = far;
    float fx = pos[far*3+0], fy = pos[far*3+1], fz = pos[far*3+2];
    float bv = -1.0f; int bi = 0x7fffffff;
    for (int j = t; j < Np; j += 1024) {
      float dx = pos[j*3+0]-fx, dy = pos[j*3+1]-fy, dz = pos[j*3+2]-fz;
      float d  = dx*dx + dy*dy + dz*dz;
      float nd = fminf(dist[j], d);
      dist[j] = nd;
      if (nd > bv) { bv = nd; bi = j; }
    }
    // wave32 shuffle argmax (tie -> lower index)
    for (int m = 16; m > 0; m >>= 1) {
      float ov = __shfl_xor(bv, m);
      int   oi = __shfl_xor(bi, m);
      if (ov > bv || (ov == bv && oi < bi)) { bv = ov; bi = oi; }
    }
    if ((t & 31) == 0) { svals[t >> 5] = bv; sidx[t >> 5] = bi; }
    __syncthreads();
    if (t == 0) {
      float vv = svals[0]; int ii = sidx[0];
      for (int w = 1; w < 32; ++w)
        if (svals[w] > vv || (svals[w] == vv && sidx[w] < ii)) { vv = svals[w]; ii = sidx[w]; }
      sFar = ii;
    }
    __syncthreads();
    far = sFar;
    __syncthreads();
  }
}

__global__ void k_gather_pos(const float* __restrict__ pos, const int* __restrict__ idx,
                             float* __restrict__ out, int n) {
  int i = blockIdx.x * blockDim.x + threadIdx.x;
  if (i < n * 3) { int p = i / 3, c = i - p * 3; out[i] = pos[idx[p]*3 + c]; }
}

// ---------------------------------------------------------------------------
// radius ball query: <=64 nearest within radius (set semantics; max-pool later)
// LDS distance array + 64 argmin rounds (wave shuffle + 8-way cross-wave)
// ---------------------------------------------------------------------------
template<int NS>
__global__ __launch_bounds__(256)
void k_select(const float* __restrict__ posSrc, const float* __restrict__ posQ,
              float r2, int* __restrict__ nbr, int* __restrict__ cnt) {
  __shared__ float sd[NS];
  __shared__ float wv[8];
  __shared__ int   wi[8];
  __shared__ float bV;
  __shared__ int   bI;
  int q = blockIdx.x;
  int t = threadIdx.x;
  float qx = posQ[q*3+0], qy = posQ[q*3+1], qz = posQ[q*3+2];
  for (int j = t; j < NS; j += 256) {
    float dx = posSrc[j*3+0]-qx, dy = posSrc[j*3+1]-qy, dz = posSrc[j*3+2]-qz;
    float d = dx*dx + dy*dy + dz*dz;
    sd[j] = (d <= r2) ? d : 3.0e38f;   // exclude out-of-radius
  }
  __syncthreads();
  int count = 0;
  for (int k = 0; k < 64; ++k) {
    float bv = 3.0e38f; int bi = 0;
    for (int j = t; j < NS; j += 256) {
      float v = sd[j];
      if (v < bv) { bv = v; bi = j; }
    }
    for (int m2 = 16; m2 > 0; m2 >>= 1) {
      float ov = __shfl_xor(bv, m2);
      int   oi = __shfl_xor(bi, m2);
      if (ov < bv || (ov == bv && oi < bi)) { bv = ov; bi = oi; }
    }
    if ((t & 31) == 0) { wv[t >> 5] = bv; wi[t >> 5] = bi; }
    __syncthreads();
    if (t == 0) {
      float vv = wv[0]; int ii = wi[0];
      for (int w = 1; w < 8; ++w)
        if (wv[w] < vv || (wv[w] == vv && wi[w] < ii)) { vv = wv[w]; ii = wi[w]; }
      bV = vv; bI = ii;
      if (vv < 3.0e38f) { sd[ii] = 3.4e38f; nbr[q*64 + k] = ii; }
    }
    __syncthreads();
    if (bV >= 3.0e38f) break;          // uniform break: no candidates left
    ++count;
  }
  if (t == 0) cnt[q] = count;
}

// ---------------------------------------------------------------------------
// build A matrix rows: [x[nbr] (Cprev), pos[nbr]-pos_q (3), zero pad to Kpad] (f16)
// ---------------------------------------------------------------------------
__global__ void k_gather_feat(const half_t* __restrict__ xPrev, int Cprev,
                              const float* __restrict__ posSrc, const float* __restrict__ posQ,
                              const int* __restrict__ nbr, const int* __restrict__ cnt,
                              int nq, half_t* __restrict__ A, int Kpad) {
  int row = blockIdx.x * blockDim.x + threadIdx.x;
  if (row >= nq * 64) return;
  int q = row >> 6, k = row & 63;
  half_t* out = A + (size_t)row * Kpad;
  if (k >= cnt[q]) { for (int c = 0; c < Kpad; ++c) out[c] = (half_t)0.0f; return; }
  int j = nbr[row];
  int c = 0;
  for (; c < Cprev; ++c) out[c] = xPrev[(size_t)j * Cprev + c];
  out[c++] = (half_t)(posSrc[j*3+0] - posQ[q*3+0]);
  out[c++] = (half_t)(posSrc[j*3+1] - posQ[q*3+1]);
  out[c++] = (half_t)(posSrc[j*3+2] - posQ[q*3+2]);
  for (; c < Kpad; ++c) out[c] = (half_t)0.0f;
}

// max over valid neighbors (invalid rows never read -> matches -inf masking)
__global__ void k_max_agg(const half_t* __restrict__ h, const int* __restrict__ cnt,
                          int nq, int C, half_t* __restrict__ out) {
  int i = blockIdx.x * blockDim.x + threadIdx.x;
  if (i >= nq * C) return;
  int q = i / C, c = i - q * C;
  int n = cnt[q]; if (n < 1) n = 1;
  const half_t* base = h + (size_t)(q * 64) * C + c;
  float m = (float)base[0];
  for (int k = 1; k < n; ++k) m = fmaxf(m, (float)base[(size_t)k * C]);
  out[(size_t)q * C + c] = (half_t)m;
}

// ---------------------------------------------------------------------------
// knn(k=3) inverse-d2 interpolation + skip concat -> A matrix (f16, Kpad)
// ---------------------------------------------------------------------------
__global__ __launch_bounds__(256)
void k_knn_interp(const float* __restrict__ posF, int Nf,
                  const float* __restrict__ posC, int Nc,
                  const half_t* __restrict__ xC, int Cc,
                  const half_t* __restrict__ xS, int Cs,
                  half_t* __restrict__ A, int Kpad) {
  __shared__ float spc[4096 * 3];
  int t = threadIdx.x;
  for (int j = t; j < Nc * 3; j += 256) spc[j] = posC[j];
  __syncthreads();
  int i = blockIdx.x * 256 + t;
  if (i >= Nf) return;
  float px = posF[i*3+0], py = posF[i*3+1], pz = posF[i*3+2];
  float d0 = 3.4e38f, d1 = 3.4e38f, d2 = 3.4e38f;
  int   i0 = 0, i1 = 0, i2 = 0;
  for (int j = 0; j < Nc; ++j) {
    float dx = spc[j*3+0]-px, dy = spc[j*3+1]-py, dz = spc[j*3+2]-pz;
    float d = dx*dx + dy*dy + dz*dz;
    if (d < d0)      { d2 = d1; i2 = i1; d1 = d0; i1 = i0; d0 = d; i0 = j; }
    else if (d < d1) { d2 = d1; i2 = i1; d1 = d;  i1 = j; }
    else if (d < d2) { d2 = d;  i2 = j; }
  }
  float w0 = 1.0f / fmaxf(d0, 1e-16f);
  float w1 = 1.0f / fmaxf(d1, 1e-16f);
  float w2 = 1.0f / fmaxf(d2, 1e-16f);
  float ws = w0 + w1 + w2; w0 /= ws; w1 /= ws; w2 /= ws;
  half_t* out = A + (size_t)i * Kpad;
  const half_t* b0 = xC + (size_t)i0 * Cc;
  const half_t* b1 = xC + (size_t)i1 * Cc;
  const half_t* b2 = xC + (size_t)i2 * Cc;
  int c = 0;
  for (; c < Cc; ++c)
    out[c] = (half_t)(w0 * (float)b0[c] + w1 * (float)b1[c] + w2 * (float)b2[c]);
  for (int s = 0; s < Cs; ++s, ++c) out[c] = xS[(size_t)i * Cs + s];
  for (; c < Kpad; ++c) out[c] = (half_t)0.0f;
}

// ---------------------------------------------------------------------------
// WMMA GEMM helpers
// ---------------------------------------------------------------------------
__device__ __forceinline__ v16h load_frag16(const half_t* __restrict__ p) {
  v8h lo = *(const v8h*)p;          // k0 + khalf*8 + [0..7]
  v8h hi = *(const v8h*)(p + 16);   // k0 + 16 + khalf*8 + [0..7]
  return __builtin_shufflevector(lo, hi, 0,1,2,3,4,5,6,7,8,9,10,11,12,13,14,15);
}

__device__ __forceinline__ v8f wmma_f16(v16h a, v16h b, v8f c) {
  return __builtin_amdgcn_wmma_f32_16x16x32_f16(false, a, false, b, (short)0, c,
                                                false, false);
}

__device__ __forceinline__ void epilogue_tile(v8f acc, int n, int mBase, int khalf,
                                              const float* __restrict__ scale,
                                              const float* __restrict__ shift, int relu,
                                              half_t* __restrict__ outH, int ldH,
                                              float* __restrict__ outF, int ldF, int Ncols) {
  float sc = scale[n], sh = shift[n];
#pragma unroll
  for (int r = 0; r < 8; ++r) {
    int m = mBase + r + 8 * khalf;
    float v = acc[r] * sc + sh;
    if (relu) v = fmaxf(v, 0.0f);
    if (outH) outH[(size_t)m * ldH + n] = (half_t)v;
    if (outF && n < Ncols) outF[(size_t)m * ldF + n] = v;
  }
}

// WMMA GEMM: out = epilogue(A[MxKpad] @ B[KpadxN])
//   A f16 row-major (Kpad mult of 32), Bt f16 transposed (N x Kpad).
//   Wave32: each wave owns a 16x(NSUB*16) output tile; NSUB is compile-time so the
//   inner loop is branch-free straight-line WMMA with in-place accumulators.
template<int NSUB>
__global__ __launch_bounds__(256)
void k_wmma_gemm(const half_t* __restrict__ A, int M, int Kpad,
                 const half_t* __restrict__ Bt,
                 const float* __restrict__ scale, const float* __restrict__ shift,
                 int relu,
                 half_t* __restrict__ outH, int ldH,
                 float* __restrict__ outF, int ldF, int Ncols) {
  int wave  = threadIdx.x >> 5;
  int lane  = threadIdx.x & 31;
  int mtile = blockIdx.x * 8 + wave;
  if (mtile * 16 >= M) return;                        // wave-uniform
  int l15   = lane & 15;
  int khalf = lane >> 4;
  int nBase = blockIdx.y * (NSUB * 16);
  int mBase = mtile * 16;
  const half_t* aP = A  + (size_t)(mBase + l15) * Kpad + khalf * 8;
  const half_t* bP = Bt + (size_t)(nBase + l15) * Kpad + khalf * 8;
  const size_t  bs = (size_t)16 * Kpad;               // stride between n-subtiles
  v8f acc0 = {0.f,0.f,0.f,0.f,0.f,0.f,0.f,0.f};
  v8f acc1 = acc0, acc2 = acc0, acc3 = acc0;
  for (int k0 = 0; k0 < Kpad; k0 += 32) {
    v16h af = load_frag16(aP + k0);
    acc0 = wmma_f16(af, load_frag16(bP + k0), acc0);
    if (NSUB > 1) acc1 = wmma_f16(af, load_frag16(bP + bs     + k0), acc1);
    if (NSUB > 2) acc2 = wmma_f16(af, load_frag16(bP + 2 * bs + k0), acc2);
    if (NSUB > 3) acc3 = wmma_f16(af, load_frag16(bP + 3 * bs + k0), acc3);
  }
  epilogue_tile(acc0, nBase + l15, mBase, khalf, scale, shift, relu, outH, ldH, outF, ldF, Ncols);
  if (NSUB > 1) epilogue_tile(acc1, nBase + 16 + l15, mBase, khalf, scale, shift, relu, outH, ldH, outF, ldF, Ncols);
  if (NSUB > 2) epilogue_tile(acc2, nBase + 32 + l15, mBase, khalf, scale, shift, relu, outH, ldH, outF, ldF, Ncols);
  if (NSUB > 3) epilogue_tile(acc3, nBase + 48 + l15, mBase, khalf, scale, shift, relu, outH, ldH, outF, ldF, Ncols);
}

// ---------------------------------------------------------------------------
// host orchestration
// ---------------------------------------------------------------------------
extern "C" void kernel_launch(void* const* d_in, const int* in_sizes, int n_in,
                              void* d_out, int out_size, void* d_ws, size_t ws_size,
                              hipStream_t stream) {
  (void)in_sizes; (void)out_size; (void)ws_size;
  const float* x_in = (const float*)d_in[0];
  const float* pos0 = (const float*)d_in[1];
  // d_in[2] = batch, unused (single cloud)

  // --- 20 linear layers (16 linbn in sa/fp mlps, sem.l1/l2, inst.l1/l2) ---
  static const int LK[20]  = {7,64, 67,128, 131,256, 259,512, 768,256, 384,128, 192,64, 68,64, 64,64, 64,64};
  static const int LN[20]  = {64,64, 128,128, 256,256, 512,512, 256,256, 128,128, 64,64, 64,64, 64,8, 64,64};
  static const int LBN[20] = {1,1, 1,1, 1,1, 1,1, 1,1, 1,1, 1,1, 1,1, 1,0, 1,0};
  int wl[20]; { int c = 0; for (int l = 0; l < 20; ++l) { wl[l] = c; c += LBN[l] ? 4 : 2; } } // 76 leaves
  int KP[20], NP[20];
  for (int l = 0; l < 20; ++l) { KP[l] = (LK[l] + 31) & ~31; NP[l] = (LN[l] + 15) & ~15; }

  // resolve param leaves: either one d_in entry per leaf, or one packed buffer
  const float* leaf[76];
  if (n_in >= 3 + 76) {
    for (int i = 0; i < 76; ++i) leaf[i] = (const float*)d_in[3 + i];
  } else {
    const float* base = (const float*)d_in[3];
    size_t off = 0; int li = 0;
    for (int l = 0; l < 20; ++l) {
      leaf[li++] = base + off; off += (size_t)LK[l] * LN[l];
      leaf[li++] = base + off; off += LN[l];
      if (LBN[l]) { leaf[li++] = base + off; off += LN[l];
                    leaf[li++] = base + off; off += LN[l]; }
    }
  }

  // --- deterministic bump allocator over d_ws ---
  size_t wsOff = 0; char* wsBase = (char*)d_ws;
  auto alloc = [&](size_t bytes) -> void* {
    size_t a = (wsOff + 255) & ~(size_t)255;
    wsOff = a + bytes;
    return (void*)(wsBase + a);
  };

  half_t* Bt[20]; float* Sc[20]; float* Sh[20];
  for (int l = 0; l < 20; ++l) {
    Bt[l] = (half_t*)alloc((size_t)NP[l] * KP[l] * sizeof(half_t));
    Sc[l] = (float*)alloc((size_t)NP[l] * sizeof(float));
    Sh[l] = (float*)alloc((size_t)NP[l] * sizeof(float));
  }
  float* distWs = (float*)alloc(16384 * sizeof(float));
  const int nQ[4] = {4096, 1024, 256, 64};
  int* idxL[4]; float* posL[4]; int* nbrL[4]; int* cntL[4];
  for (int v = 0; v < 4; ++v) {
    idxL[v] = (int*)alloc((size_t)nQ[v] * 4);
    posL[v] = (float*)alloc((size_t)nQ[v] * 3 * 4);
    nbrL[v] = (int*)alloc((size_t)nQ[v] * 64 * 4);
    cntL[v] = (int*)alloc((size_t)nQ[v] * 4);
  }
  half_t* xl0  = (half_t*)alloc((size_t)16384 * 4 * 2);
  half_t* x1   = (half_t*)alloc((size_t)4096 * 64 * 2);
  half_t* x2   = (half_t*)alloc((size_t)1024 * 128 * 2);
  half_t* x3   = (half_t*)alloc((size_t)256 * 256 * 2);
  half_t* x4   = (half_t*)alloc((size_t)64 * 512 * 2);
  half_t* dd3  = (half_t*)alloc((size_t)256 * 256 * 2);
  half_t* dd2  = (half_t*)alloc((size_t)1024 * 128 * 2);
  half_t* dd1  = (half_t*)alloc((size_t)4096 * 64 * 2);
  half_t* feat = (half_t*)alloc((size_t)16384 * 64 * 2);
  half_t* bufA = (half_t*)alloc((size_t)262144 * 64 * 2);   // 32 MB
  half_t* bufB = (half_t*)alloc((size_t)262144 * 64 * 2);   // 32 MB

  // --- pack weights + convert input features to f16 ---
  for (int l = 0; l < 20; ++l) {
    const float* W  = leaf[wl[l]];
    const float* b  = leaf[wl[l] + 1];
    const float* g  = LBN[l] ? leaf[wl[l] + 2] : b;
    const float* be = LBN[l] ? leaf[wl[l] + 3] : b;
    int total = NP[l] * KP[l]; if (total < NP[l]) total = NP[l];
    k_pack_linbn<<<dim3((total + 255) / 256), dim3(256), 0, stream>>>(
        W, b, g, be, LK[l], LN[l], KP[l], NP[l], LBN[l], Bt[l], Sc[l], Sh[l]);
  }
  k_f32_to_f16<<<dim3((16384 * 4 + 255) / 256), dim3(256), 0, stream>>>(x_in, xl0, 16384 * 4);

  auto gemm = [&](const half_t* Ain, int M, int Kp, int l,
                  half_t* oH, float* oF, int ldF, int Ncols) {
    if ((NP[l] & 63) == 0) {
      dim3 g((M + 127) / 128, NP[l] / 64);
      k_wmma_gemm<4><<<g, dim3(256), 0, stream>>>(Ain, M, Kp, Bt[l], Sc[l], Sh[l],
                                                  LBN[l], oH, NP[l], oF, ldF, Ncols);
    } else {
      dim3 g((M + 127) / 128, NP[l] / 16);
      k_wmma_gemm<1><<<g, dim3(256), 0, stream>>>(Ain, M, Kp, Bt[l], Sc[l], Sh[l],
                                                  LBN[l], oH, NP[l], oF, ldF, Ncols);
    }
  };

  auto select = [&](int Ns, const float* posSrc, const float* posQ, float r2,
                    int* nbr, int* cnt, int nq) {
    dim3 g(nq), bl(256);
    switch (Ns) {
      case 16384: k_select<16384><<<g, bl, 0, stream>>>(posSrc, posQ, r2, nbr, cnt); break;
      case 4096:  k_select<4096> <<<g, bl, 0, stream>>>(posSrc, posQ, r2, nbr, cnt); break;
      case 1024:  k_select<1024> <<<g, bl, 0, stream>>>(posSrc, posQ, r2, nbr, cnt); break;
      default:    k_select<256>  <<<g, bl, 0, stream>>>(posSrc, posQ, r2, nbr, cnt); break;
    }
  };

  auto runSA = [&](const float* posSrc, int Ns, int lvl, const half_t* xPrev, int Cprev,
                   float r, int L0, half_t* xOut) {
    int nq = nQ[lvl];
    k_fps<<<dim3(1), dim3(1024), 0, stream>>>(posSrc, Ns, nq, idxL[lvl], distWs);
    k_gather_pos<<<dim3((nq * 3 + 255) / 256), dim3(256), 0, stream>>>(posSrc, idxL[lvl], posL[lvl], nq);
    select(Ns, posSrc, posL[lvl], r * r, nbrL[lvl], cntL[lvl], nq);
    int M = nq * 64;
    k_gather_feat<<<dim3((M + 255) / 256), dim3(256), 0, stream>>>(
        xPrev, Cprev, posSrc, posL[lvl], nbrL[lvl], cntL[lvl], nq, bufA, KP[L0]);
    gemm(bufA, M, KP[L0], L0,     bufB, nullptr, 0, 0);
    gemm(bufB, M, NP[L0], L0 + 1, bufA, nullptr, 0, 0);
    int C2 = LN[L0 + 1];
    k_max_agg<<<dim3((nq * C2 + 255) / 256), dim3(256), 0, stream>>>(bufA, cntL[lvl], nq, C2, xOut);
  };

  auto runFP = [&](const float* posF, int Nf, const float* posC, int Nc,
                   const half_t* xC, int Cc, const half_t* xS, int Cs,
                   int L0, half_t* dOut) {
    k_knn_interp<<<dim3((Nf + 255) / 256), dim3(256), 0, stream>>>(
        posF, Nf, posC, Nc, xC, Cc, xS, Cs, bufA, KP[L0]);
    gemm(bufA, Nf, KP[L0], L0,     bufB, nullptr, 0, 0);
    gemm(bufB, Nf, NP[L0], L0 + 1, dOut, nullptr, 0, 0);
  };

  // --- encoder (SA modules) ---
  runSA(pos0,    16384, 0, xl0, 4,   0.1f, 0, x1);
  runSA(posL[0], 4096,  1, x1,  64,  0.2f, 2, x2);
  runSA(posL[1], 1024,  2, x2,  128, 0.4f, 4, x3);
  runSA(posL[2], 256,   3, x3,  256, 0.8f, 6, x4);

  // --- decoder (FP modules) ---
  runFP(posL[2], 256,   posL[3], 64,   x4,  512, x3,  256, 8,  dd3);
  runFP(posL[1], 1024,  posL[2], 256,  dd3, 256, x2,  128, 10, dd2);
  runFP(posL[0], 4096,  posL[1], 1024, dd2, 128, x1,  64,  12, dd1);
  runFP(pos0,    16384, posL[0], 4096, dd1, 64,  xl0, 4,   14, feat);

  // --- heads (sem [16384x8] then inst [16384x64], f32 into d_out) ---
  float* semOut  = (float*)d_out;
  float* instOut = semOut + (size_t)16384 * 8;
  gemm(feat, 16384, 64,     16, bufB, nullptr, 0, 0);
  gemm(bufB, 16384, NP[16], 17, nullptr, semOut, 8, 8);
  gemm(feat, 16384, 64,     18, bufB, nullptr, 0, 0);
  gemm(bufB, 16384, NP[18], 19, nullptr, instOut, 64, 64);
}
